// Backbone_24781961298012
// MI455X (gfx1250) — matrix operand order repaired and checked by reading, not compiled, not run on hardware
//
#include <hip/hip_runtime.h>
#include <math.h>

#define KNN 16
#define NB 2
#define DM 512
#define TM 32
#define TN 64
#define TKK 32
#define FTM 64
#define FTN 128
#define CHUNK_ROWS 1024

typedef __attribute__((ext_vector_type(16))) _Float16 v16h;
typedef __attribute__((ext_vector_type(8)))  _Float16 v8h;
typedef __attribute__((ext_vector_type(4)))  _Float16 v4h;
typedef __attribute__((ext_vector_type(8)))  float    v8f;

// ============================================================================
// FAST WMMA GEMM (M%64==0, N%128==0, Kd%32==0), LDS double-buffered:
// 256 threads = 8 wave32 waves (2x4). Block tile 64x128; each wave owns a
// 32x32 subtile = 4 v_wmma_f32_16x16x32_f16 per K step. Global loads for
// tile k+1 issue BEFORE tile k's WMMAs; ping-pong LDS -> 1 barrier per step.
// ============================================================================
__global__ __launch_bounds__(256)
void gemm_wmma_fast(const float* __restrict__ A, const float* __restrict__ W,
                    const float* __restrict__ bias, const float* __restrict__ resid,
                    float* __restrict__ C, int M, int N, int Kd, int relu)
{
    __shared__ _Float16 As[2][FTM][TKK + 8] __attribute__((aligned(16)));
    __shared__ _Float16 Bs[2][FTN][TKK + 8] __attribute__((aligned(16)));

    const int tid   = threadIdx.x;
    const int lane  = tid & 31;
    const int wave  = tid >> 5;
    const int wm    = wave >> 2;      // 0..1 : 32 rows
    const int wn    = wave & 3;       // 0..3 : 32 cols
    const int laneM = lane & 15;
    const int laneH = lane >> 4;

    const int m0 = blockIdx.x * FTM;
    const int n0 = blockIdx.y * FTN;

    // staging coordinates
    const int ar = tid >> 3;          // 0..31 (A row, +32 on 2nd pass)
    const int ac = (tid & 7) * 4;     // A col (float4)
    const int bk = tid >> 3;          // 0..31 (W k-row)
    const int bn = (tid & 7) * 4;     // W n base (float4, stride 32)

    const int nk = Kd / TKK;

    float4 aReg[2];
    float4 wReg[4];

    auto gload = [&](int kt) {
        const int k0 = kt * TKK;
        #pragma unroll
        for (int p = 0; p < 2; ++p)
            aReg[p] = *(const float4*)&A[(size_t)(m0 + ar + 32 * p) * Kd + k0 + ac];
        #pragma unroll
        for (int p = 0; p < 4; ++p)
            wReg[p] = *(const float4*)&W[(size_t)(k0 + bk) * N + n0 + bn + 32 * p];
    };
    auto lstore = [&](int buf) {
        #pragma unroll
        for (int p = 0; p < 2; ++p) {
            v4h h = { (_Float16)aReg[p].x, (_Float16)aReg[p].y,
                      (_Float16)aReg[p].z, (_Float16)aReg[p].w };
            *(v4h*)&As[buf][ar + 32 * p][ac] = h;
        }
        #pragma unroll
        for (int p = 0; p < 4; ++p) {
            int n = bn + 32 * p;
            Bs[buf][n + 0][bk] = (_Float16)wReg[p].x;
            Bs[buf][n + 1][bk] = (_Float16)wReg[p].y;
            Bs[buf][n + 2][bk] = (_Float16)wReg[p].z;
            Bs[buf][n + 3][bk] = (_Float16)wReg[p].w;
        }
    };

    v8f acc[2][2] = {};

    gload(0);
    lstore(0);
    __syncthreads();

    for (int kt = 0; kt < nk; ++kt) {
        const int cur = kt & 1;
        // issue next tile's global loads early (latency hidden behind WMMAs)
        if (kt + 1 < nk) gload(kt + 1);

        #pragma unroll
        for (int i = 0; i < 2; ++i) {
            // A fragment: lane=M, k = 16*laneH + elem (matches B)
            const _Float16* ap = &As[cur][wm * 32 + i * 16 + laneM][laneH * 16];
            v8h a0 = *(const v8h*)(ap);
            v8h a1 = *(const v8h*)(ap + 8);
            v16h af = __builtin_shufflevector(a0, a1, 0,1,2,3,4,5,6,7,8,9,10,11,12,13,14,15);
            #pragma unroll
            for (int j = 0; j < 2; ++j) {
                // B fragment: lane=N col, k = 16*laneH + elem
                const _Float16* bp = &Bs[cur][wn * 32 + j * 16 + laneM][laneH * 16];
                v8h b0 = *(const v8h*)(bp);
                v8h b1 = *(const v8h*)(bp + 8);
                v16h bf = __builtin_shufflevector(b0, b1, 0,1,2,3,4,5,6,7,8,9,10,11,12,13,14,15);
                acc[i][j] = __builtin_amdgcn_wmma_f32_16x16x32_f16(
                    false, af, false, bf, (short)0, acc[i][j], false, false);
            }
        }

        if (kt + 1 < nk) {
            // write buffer (kt+1)&1: its last readers ran at step kt-1 and all
            // waves passed that step's barrier already -> safe with 1 barrier.
            lstore((kt + 1) & 1);
            __syncthreads();
        }
    }

    // epilogue: C/D layout m = r + 8*laneH, n = laneM (no bounds: exact tiles)
    #pragma unroll
    for (int i = 0; i < 2; ++i) {
        #pragma unroll
        for (int j = 0; j < 2; ++j) {
            int nOut = n0 + wn * 32 + j * 16 + laneM;
            float bvv = bias ? bias[nOut] : 0.0f;
            #pragma unroll
            for (int r = 0; r < 8; ++r) {
                int mOut = m0 + wm * 32 + i * 16 + r + 8 * laneH;
                float v = acc[i][j][r] + bvv;
                if (relu)  v = fmaxf(v, 0.0f);
                if (resid) v += resid[(size_t)mOut * N + nOut];
                C[(size_t)mOut * N + nOut] = v;
            }
        }
    }
}

// ============================================================================
// EDGE WMMA GEMM (ragged M/N/K): block tile 32x64, 1 WMMA per wave per K step.
// Staging is branchless (clamped index + select) so loads pipeline.
// ============================================================================
__global__ __launch_bounds__(256)
void gemm_wmma_edge(const float* __restrict__ A, const float* __restrict__ W,
                    const float* __restrict__ bias, const float* __restrict__ resid,
                    float* __restrict__ C, int M, int N, int Kd, int relu)
{
    __shared__ _Float16 As[TM][TKK + 8] __attribute__((aligned(16)));
    __shared__ _Float16 Bs[TN][TKK + 8] __attribute__((aligned(16)));

    const int tid   = threadIdx.x;
    const int lane  = tid & 31;
    const int wave  = tid >> 5;
    const int wm    = wave >> 2;
    const int wn    = wave & 3;
    const int laneM = lane & 15;
    const int laneH = lane >> 4;

    const int m0 = blockIdx.x * TM;
    const int n0 = blockIdx.y * TN;

    v8f acc = {};

    for (int k0 = 0; k0 < Kd; k0 += TKK) {
        #pragma unroll
        for (int i = tid; i < TM * TKK; i += 256) {
            int r = i >> 5, k = i & 31;
            int gm = m0 + r, gk = k0 + k;
            bool ok = (gm < M) && (gk < Kd);
            size_t src = ok ? ((size_t)gm * Kd + gk) : 0;
            float v = A[src];                         // unconditional load
            As[r][k] = (_Float16)(ok ? v : 0.0f);
        }
        #pragma unroll
        for (int i = tid; i < TN * TKK; i += 256) {
            int n = i >> 5, k = i & 31;
            int gn = n0 + n, gk = k0 + k;
            bool ok = (gn < N) && (gk < Kd);
            size_t src = ok ? ((size_t)gk * N + gn) : 0;
            float v = W[src];
            Bs[n][k] = (_Float16)(ok ? v : 0.0f);
        }
        __syncthreads();

        const _Float16* ap = &As[wm * 16 + laneM][laneH * 16];
        const _Float16* bp = &Bs[wn * 16 + laneM][laneH * 16];
        v8h a0 = *(const v8h*)(ap);
        v8h a1 = *(const v8h*)(ap + 8);
        v8h b0 = *(const v8h*)(bp);
        v8h b1 = *(const v8h*)(bp + 8);
        v16h af = __builtin_shufflevector(a0, a1, 0,1,2,3,4,5,6,7,8,9,10,11,12,13,14,15);
        v16h bf = __builtin_shufflevector(b0, b1, 0,1,2,3,4,5,6,7,8,9,10,11,12,13,14,15);
        acc = __builtin_amdgcn_wmma_f32_16x16x32_f16(false, af, false, bf,
                                                     (short)0, acc, false, false);
        __syncthreads();
    }

    const int nOut = n0 + wn * 16 + laneM;
    #pragma unroll
    for (int r = 0; r < 8; ++r) {
        int mOut = m0 + wm * 16 + r + 8 * laneH;
        if (mOut < M && nOut < N) {
            float v = acc[r];
            if (bias)  v += bias[nOut];
            if (relu)  v = fmaxf(v, 0.0f);
            if (resid) v += resid[(size_t)mOut * N + nOut];
            C[(size_t)mOut * N + nOut] = v;
        }
    }
}

// ============================================================================
// KNN: one thread per query, 16-deep insertion-sorted register list
// ============================================================================
__global__ void knn_kernel(const float* __restrict__ qxyz, const float* __restrict__ pxyz,
                           int* __restrict__ idx, int B, int Nq, int Np)
{
    int t = blockIdx.x * blockDim.x + threadIdx.x;
    if (t >= B * Nq) return;
    int b = t / Nq;
    float qx = qxyz[(size_t)t*3], qy = qxyz[(size_t)t*3+1], qz = qxyz[(size_t)t*3+2];
    const float* P = pxyz + (size_t)b * Np * 3;
    float best[KNN]; int bi[KNN];
    #pragma unroll
    for (int i = 0; i < KNN; ++i) { best[i] = 3.0e38f; bi[i] = 0; }
    for (int j = 0; j < Np; ++j) {
        float dx = qx - P[3*j], dy = qy - P[3*j+1], dz = qz - P[3*j+2];
        float d = dx*dx + dy*dy + dz*dz;
        if (d < best[KNN-1]) {
            best[KNN-1] = d; bi[KNN-1] = j;
            #pragma unroll
            for (int s = KNN-1; s > 0; --s) {
                if (best[s] < best[s-1]) {
                    float tv = best[s]; best[s] = best[s-1]; best[s-1] = tv;
                    int ti = bi[s]; bi[s] = bi[s-1]; bi[s-1] = ti;
                }
            }
        }
    }
    #pragma unroll
    for (int i = 0; i < KNN; ++i) idx[(size_t)t * KNN + i] = bi[i];
}

// ============================================================================
// Farthest point sampling: one block per batch; points + min-dist in LDS.
// ============================================================================
__global__ __launch_bounds__(256)
void fps_kernel(const float* __restrict__ xyz, int* __restrict__ out, int N, int npoint)
{
    __shared__ float sx[2048], sy[2048], sz[2048], sdist[2048];
    __shared__ float rv[256]; __shared__ int ri[256];
    int b = blockIdx.x;
    const float* P = xyz + (size_t)b * N * 3;
    for (int j = threadIdx.x; j < N; j += 256) {
        sx[j] = P[3*j]; sy[j] = P[3*j+1]; sz[j] = P[3*j+2];
        sdist[j] = 1.0e10f;
    }
    __syncthreads();
    int far = 0;
    for (int it = 0; it < npoint; ++it) {
        if (threadIdx.x == 0) out[b * npoint + it] = far;
        float cx = sx[far], cy = sy[far], cz = sz[far];
        float lbest = -1.0f; int lidx = 0;
        for (int j = threadIdx.x; j < N; j += 256) {
            float dx = sx[j]-cx, dy = sy[j]-cy, dz = sz[j]-cz;
            float nd = fminf(sdist[j], dx*dx + dy*dy + dz*dz);
            sdist[j] = nd;
            if (nd > lbest) { lbest = nd; lidx = j; }
        }
        rv[threadIdx.x] = lbest; ri[threadIdx.x] = lidx;
        __syncthreads();
        for (int s = 128; s > 0; s >>= 1) {
            if (threadIdx.x < (unsigned)s) {
                float v2 = rv[threadIdx.x + s]; int i2 = ri[threadIdx.x + s];
                if (v2 > rv[threadIdx.x] ||
                    (v2 == rv[threadIdx.x] && i2 < ri[threadIdx.x])) {
                    rv[threadIdx.x] = v2; ri[threadIdx.x] = i2;
                }
            }
            __syncthreads();
        }
        far = ri[0];
        __syncthreads();
    }
}

__global__ void extract_xyz_kernel(const float* __restrict__ in, float* __restrict__ out, int rows)
{
    size_t t = (size_t)blockIdx.x * blockDim.x + threadIdx.x;
    if (t >= (size_t)rows * 3) return;
    int r = (int)(t / 3), c = (int)(t % 3);
    out[t] = in[(size_t)r * 5 + c];
}

__global__ void gather_xyz_kernel(const float* __restrict__ xyz, const int* __restrict__ fidx,
                                  float* __restrict__ out, int N, int np)
{
    size_t t = (size_t)blockIdx.x * blockDim.x + threadIdx.x;
    if (t >= (size_t)NB * np) return;
    int b = (int)(t / np);
    int s = fidx[t];
    const float* p = xyz + ((size_t)b * N + s) * 3;
    out[t*3+0] = p[0]; out[t*3+1] = p[1]; out[t*3+2] = p[2];
}

__global__ void build_rel_kernel(const float* __restrict__ xyz, const int* __restrict__ idx,
                                 float* __restrict__ rel, int r0, int rows, int N)
{
    size_t t = (size_t)blockIdx.x * blockDim.x + threadIdx.x;
    if (t >= (size_t)rows * KNN) return;
    int rr = (int)(t / KNN), k = (int)(t % KNN);
    int r = r0 + rr, b = r / N;
    int j = idx[(size_t)r * KNN + k];
    const float* pq = xyz + (size_t)r * 3;
    const float* pn = xyz + ((size_t)b * N + j) * 3;
    rel[t*3+0] = pq[0]-pn[0]; rel[t*3+1] = pq[1]-pn[1]; rel[t*3+2] = pq[2]-pn[2];
}

__global__ void build_t_kernel(const float* __restrict__ q, const float* __restrict__ kk,
                               const float* __restrict__ pos, const int* __restrict__ idx,
                               float* __restrict__ t, int r0, int rows, int N)
{
    size_t e = (size_t)blockIdx.x * blockDim.x + threadIdx.x;
    if (e >= (size_t)rows * KNN * DM) return;
    int f = (int)(e & (DM - 1));
    size_t rk = e >> 9;
    int rr = (int)(rk >> 4), k = (int)(rk & 15);
    int r = r0 + rr, b = r / N;
    int j = idx[(size_t)r * KNN + k];
    t[e] = q[(size_t)r * DM + f] - kk[((size_t)b * N + j) * DM + f] + pos[e];
}

__global__ void softmax_ws_kernel(const float* __restrict__ attn, const float* __restrict__ pos,
                                  const float* __restrict__ vall, const int* __restrict__ idx,
                                  float* __restrict__ res, int r0, int rows, int N)
{
    size_t e = (size_t)blockIdx.x * blockDim.x + threadIdx.x;
    if (e >= (size_t)rows * DM) return;
    int f = (int)(e & (DM - 1));
    int rr = (int)(e >> 9);
    int r = r0 + rr, b = r / N;
    const float scale = 0.04419417382415922f;   // 1/sqrt(512)
    float a[KNN], mx = -3.0e38f;
    #pragma unroll
    for (int k = 0; k < KNN; ++k) {
        a[k] = attn[((size_t)rr * KNN + k) * DM + f] * scale;
        mx = fmaxf(mx, a[k]);
    }
    float s = 0.0f;
    #pragma unroll
    for (int k = 0; k < KNN; ++k) { a[k] = expf(a[k] - mx); s += a[k]; }
    float inv = 1.0f / s, acc = 0.0f;
    #pragma unroll
    for (int k = 0; k < KNN; ++k) {
        int j = idx[(size_t)r * KNN + k];
        float vv = vall[((size_t)b * N + j) * DM + f] + pos[((size_t)rr * KNN + k) * DM + f];
        acc += a[k] * inv * vv;
    }
    res[(size_t)r * DM + f] = acc;
}

__global__ void build_group_kernel(const float* __restrict__ xyz, const float* __restrict__ nxyz,
                                   const float* __restrict__ feats, const int* __restrict__ idx,
                                   float* __restrict__ g, int N, int np, int Fin)
{
    int W = Fin + 3;
    size_t e = (size_t)blockIdx.x * blockDim.x + threadIdx.x;
    if (e >= (size_t)NB * np * KNN * W) return;
    int c = (int)(e % W);
    size_t rk = e / W;
    int k  = (int)(rk % KNN);
    int qi = (int)(rk / KNN);
    int b  = qi / np;
    int j  = idx[(size_t)qi * KNN + k];
    float v;
    if (c < 3) v = xyz[((size_t)b * N + j) * 3 + c] - nxyz[(size_t)qi * 3 + c];
    else       v = feats[((size_t)b * N + j) * Fin + (c - 3)];
    g[e] = v;
}

__global__ void maxpool_kernel(const float* __restrict__ h, float* __restrict__ out,
                               int rows, int C)
{
    size_t t = (size_t)blockIdx.x * blockDim.x + threadIdx.x;
    if (t >= (size_t)rows * C) return;
    int c = (int)(t % C), q = (int)(t / C);
    float m = -3.0e38f;
    #pragma unroll
    for (int k = 0; k < KNN; ++k)
        m = fmaxf(m, h[((size_t)q * KNN + k) * C + c]);
    out[t] = m;
}

// ============================================================================
extern "C" void kernel_launch(void* const* d_in, const int* in_sizes, int n_in,
                              void* d_out, int out_size, void* d_ws, size_t ws_size,
                              hipStream_t stream)
{
    (void)in_sizes; (void)n_in; (void)out_size; (void)ws_size;
    auto F = [&](int i) { return (const float*)d_in[i]; };
    float* out = (float*)d_out;

    char* wsb = (char*)d_ws; size_t off = 0;
    auto allocF = [&](size_t n) { float* p = (float*)(wsb + off);
        off += ((n * sizeof(float) + 255) / 256) * 256; return p; };
    auto allocI = [&](size_t n) { int* p = (int*)(wsb + off);
        off += ((n * sizeof(int) + 255) / 256) * 256; return p; };

    float* xyz0   = allocF((size_t)NB*2048*3);
    float* feats0 = allocF((size_t)NB*2048*32);
    float* xyz1   = allocF((size_t)NB*512*3);
    float* feats1 = allocF((size_t)NB*512*64);
    float* xyz2   = allocF((size_t)NB*128*3);
    float* feats2 = allocF((size_t)NB*128*128);
    float* bx     = allocF((size_t)NB*2048*DM);
    float* bq     = allocF((size_t)NB*2048*DM);
    float* bk     = allocF((size_t)NB*2048*DM);
    float* bv     = allocF((size_t)NB*2048*DM);
    float* bres   = allocF((size_t)NB*2048*DM);
    float* brel   = allocF((size_t)CHUNK_ROWS*KNN*3);
    float* bA     = allocF((size_t)CHUNK_ROWS*KNN*DM);
    float* bP     = allocF((size_t)CHUNK_ROWS*KNN*DM);
    float* bC     = allocF((size_t)CHUNK_ROWS*KNN*DM);
    float* bg     = allocF((size_t)NB*512*KNN*67);
    float* bh1    = allocF((size_t)NB*512*KNN*64);
    float* bh2    = allocF((size_t)NB*512*KNN*64);
    int*   bidx   = allocI((size_t)NB*2048*KNN);
    int*   bfps   = allocI((size_t)NB*512);

    auto gemm = [&](const float* A, const float* W, const float* b, const float* resid,
                    float* C, int M, int N, int Kd, int relu) {
        if ((M % FTM) == 0 && (N % FTN) == 0 && (Kd % TKK) == 0) {
            dim3 grid(M / FTM, N / FTN);
            gemm_wmma_fast<<<grid, 256, 0, stream>>>(A, W, b, resid, C, M, N, Kd, relu);
        } else {
            dim3 grid((M + TM - 1) / TM, (N + TN - 1) / TN);
            gemm_wmma_edge<<<grid, 256, 0, stream>>>(A, W, b, resid, C, M, N, Kd, relu);
        }
    };
    auto g1 = [](size_t n) { return dim3((unsigned)((n + 255) / 256)); };

    // Param indices: jax tree_flatten order (dict keys sorted).
    // 0 input | 1-4 fc1 stem (b,w,b,w) | 5-8 stage0.td | 9-23 stage0.tf
    // 24-27 stage1.td | 28-42 stage1.tf | 43-57 transformer1
    // tf base+: 0 alpha.w 1 d0.b 2 d0.w 3 d1.b 4 d1.w 5 fc1.b 6 fc1.w
    //           7 fc2.b 8 fc2.w 9 g0.b 10 g0.w 11 g1.b 12 g1.w 13 phi.w 14 psi.w

    auto pt_block = [&](int base, const float* xyz, float* feats, int N, int d) {
        int R = NB * N;
        knn_kernel<<<g1((size_t)R), 256, 0, stream>>>(xyz, xyz, bidx, NB, N, N);
        gemm(feats, F(base+6), F(base+5), nullptr, bx, R, DM, d, 0);   // fc1
        gemm(bx, F(base+13), nullptr, nullptr, bq, R, DM, DM, 0);      // phi
        gemm(bx, F(base+14), nullptr, nullptr, bk, R, DM, DM, 0);      // psi
        gemm(bx, F(base+0),  nullptr, nullptr, bv, R, DM, DM, 0);      // alpha
        for (int r0 = 0; r0 < R; r0 += CHUNK_ROWS) {
            int rows = (R - r0 < CHUNK_ROWS) ? (R - r0) : CHUNK_ROWS;
            int MK = rows * KNN;
            build_rel_kernel<<<g1((size_t)MK), 256, 0, stream>>>(xyz, bidx, brel, r0, rows, N);
            gemm(brel, F(base+2), F(base+1), nullptr, bA, MK, DM, 3, 1);    // delta0+relu
            gemm(bA,   F(base+4), F(base+3), nullptr, bP, MK, DM, DM, 0);   // delta1 -> pos
            build_t_kernel<<<g1((size_t)MK*DM), 256, 0, stream>>>(bq, bk, bP, bidx, bA, r0, rows, N);
            gemm(bA, F(base+10), F(base+9),  nullptr, bC, MK, DM, DM, 1);   // gamma0+relu
            gemm(bC, F(base+12), F(base+11), nullptr, bA, MK, DM, DM, 0);   // gamma1 -> attn
            softmax_ws_kernel<<<g1((size_t)rows*DM), 256, 0, stream>>>(bA, bP, bv, bidx, bres, r0, rows, N);
        }
        gemm(bres, F(base+8), F(base+7), feats, feats, R, d, DM, 0);   // fc2 + residual
    };

    auto trans_down = [&](const float* xyzi, const float* featsi, float* xyzo, float* featso,
                          int N, int Fin, const float* t0w, const float* t0b,
                          const float* t1w, const float* t1b) {
        int np = N / 4, Co = 2 * Fin, W = Fin + 3;
        fps_kernel<<<dim3(NB), 256, 0, stream>>>(xyzi, bfps, N, np);
        gather_xyz_kernel<<<g1((size_t)NB*np), 256, 0, stream>>>(xyzi, bfps, xyzo, N, np);
        knn_kernel<<<g1((size_t)NB*np), 256, 0, stream>>>(xyzo, xyzi, bidx, NB, np, N);
        build_group_kernel<<<g1((size_t)NB*np*KNN*W), 256, 0, stream>>>(xyzi, xyzo, featsi, bidx, bg, N, np, Fin);
        int MK = NB * np * KNN;
        gemm(bg,  t0w, t0b, nullptr, bh1, MK, Co, W,  1);
        gemm(bh1, t1w, t1b, nullptr, bh2, MK, Co, Co, 1);
        maxpool_kernel<<<g1((size_t)NB*np*Co), 256, 0, stream>>>(bh2, featso, NB*np, Co);
    };

    // stem: (B*2048,5) -> 32 (relu) -> 32
    extract_xyz_kernel<<<g1((size_t)NB*2048*3), 256, 0, stream>>>((const float*)d_in[0], xyz0, NB*2048);
    gemm((const float*)d_in[0], F(2), F(1), nullptr, bx, NB*2048, 32, 5, 1);
    gemm(bx, F(4), F(3), nullptr, feats0, NB*2048, 32, 32, 0);

    pt_block(43, xyz0, feats0, 2048, 32);
    trans_down(xyz0, feats0, xyz1, feats1, 2048, 32, F(6), F(5), F(8), F(7));
    pt_block(9, xyz1, feats1, 512, 64);
    trans_down(xyz1, feats1, xyz2, feats2, 512, 64, F(25), F(24), F(27), F(26));
    pt_block(28, xyz2, feats2, 128, 128);

    // output: (points, [(xyz0,f0),(xyz1,f1),(xyz2,f2)]) flattened
    hipMemcpyAsync(out,          feats2, sizeof(float)*32768,  hipMemcpyDeviceToDevice, stream);
    hipMemcpyAsync(out+32768,    xyz0,   sizeof(float)*12288,  hipMemcpyDeviceToDevice, stream);
    hipMemcpyAsync(out+45056,    feats0, sizeof(float)*131072, hipMemcpyDeviceToDevice, stream);
    hipMemcpyAsync(out+176128,   xyz1,   sizeof(float)*3072,   hipMemcpyDeviceToDevice, stream);
    hipMemcpyAsync(out+179200,   feats1, sizeof(float)*65536,  hipMemcpyDeviceToDevice, stream);
    hipMemcpyAsync(out+244736,   xyz2,   sizeof(float)*768,    hipMemcpyDeviceToDevice, stream);
    hipMemcpyAsync(out+245504,   feats2, sizeof(float)*32768,  hipMemcpyDeviceToDevice, stream);
}
